// GridSpatialEncoder_5540507812265
// MI455X (gfx1250) — matrix operand order repaired and checked by reading, not compiled, not run on hardware
//
#include <hip/hip_runtime.h>
#include <hip/hip_bf16.h>

typedef __attribute__((ext_vector_type(16))) __bf16          v16bf;
typedef __attribute__((ext_vector_type(8)))  float           v8f;
typedef __attribute__((ext_vector_type(4)))  float           v4f;
typedef __attribute__((ext_vector_type(8)))  unsigned short  u16x8;
typedef __attribute__((ext_vector_type(16))) unsigned short  u16x16;

static constexpr int Bn   = 4;
static constexpr int Np   = 16384;
static constexpr int FINc = 128;
static constexpr int Mc   = 1024;       // 32x32 grid cells
static constexpr int ROWS = Bn * Np;    // 65536 points

#define DEVFN __device__ __forceinline__

DEVFN unsigned short f2bf(float f) {               // RNE float -> bf16
    unsigned int u = __float_as_uint(f);
    u += 0x7fffu + ((u >> 16) & 1u);
    return (unsigned short)(u >> 16);
}

union VB { u16x16 u; v16bf b; };

// ---- WMMA fragment loaders (ISA 7.12.2 layouts) -----------------------------
// A (16x32 bf16): lane = half*16 + m; element e: k = kBase + (e<8 ? half*8+e
//                                                           : 16+half*8+(e-8))
DEVFN v16bf load_a_bf16(const unsigned short* A, int lda, int rowBase, int kBase, int lane) {
    int half = lane >> 4, m = lane & 15;
    const unsigned short* p = A + (size_t)(rowBase + m) * lda + kBase + half * 8;
    u16x8 lo = *(const u16x8*)p;
    u16x8 hi = *(const u16x8*)(p + 16);
    VB r;
#pragma unroll
    for (int i = 0; i < 8; ++i) { r.u[i] = lo[i]; r.u[8 + i] = hi[i]; }
    return r.b;
}

DEVFN v16bf load_a_f32(const float* A, int lda, int rowBase, int kBase, int lane) {
    int half = lane >> 4, m = lane & 15;
    const float* p = A + (size_t)(rowBase + m) * lda + kBase + half * 8;
    v4f a0 = *(const v4f*)p,        a1 = *(const v4f*)(p + 4);
    v4f b0 = *(const v4f*)(p + 16), b1 = *(const v4f*)(p + 20);
    VB r;
#pragma unroll
    for (int i = 0; i < 4; ++i) {
        r.u[i]      = f2bf(a0[i]);
        r.u[4 + i]  = f2bf(a1[i]);
        r.u[8 + i]  = f2bf(b0[i]);
        r.u[12 + i] = f2bf(b1[i]);
    }
    return r.b;
}

// B (32x16 bf16) packed so each lane's 16 values are one contiguous 32B chunk
DEVFN v16bf load_b(const unsigned short* Bp, int kt, int nt, int lane) {
    VB r;
    r.u = ((const u16x16*)Bp)[(size_t)(kt * 16 + nt) * 32 + lane];
    return r.b;
}

// ---- weight packer: row-major f32 (Kdim x N) -> bf16 B-fragment layout ------
__global__ void pack_w(const float* __restrict__ W, unsigned short* __restrict__ Wp,
                       int Kdim, int N) {
    int idx = blockIdx.x * blockDim.x + threadIdx.x;
    if (idx >= Kdim * N) return;
    int e = idx & 15, lane = (idx >> 4) & 31, f = idx >> 9;
    int NT = N >> 4;
    int kt = f / NT, nt = f % NT;
    int half = lane >> 4, n = lane & 15;
    int k = kt * 32 + (e < 8 ? half * 8 + e : 16 + half * 8 + (e - 8));
    int col = nt * 16 + n;
    Wp[idx] = f2bf(W[(size_t)k * N + col]);
}

// ---- WMMA GEMM: C(rows x 256) = A(rows x KDIM) @ B + bias ------------------
// Wave tile: 16 rows x 128 cols (acc[8] = 64 VGPRs -> no scratch spills).
// blockIdx.y in {0,1} selects the column half.
template <int KDIM, bool AF32, bool BIAS>
__global__ void __launch_bounds__(256)
gemm256(const void* __restrict__ Aptr, const unsigned short* __restrict__ Bp,
        const float* __restrict__ bias, float* __restrict__ C, int rows) {
    int wave = blockIdx.x * (blockDim.x >> 5) + (threadIdx.x >> 5);
    int lane = threadIdx.x & 31;
    int rowBase = wave << 4;
    int ntBase = blockIdx.y * 8;
    if (rowBase >= rows) return;
    v8f acc[8] = {};
    for (int kt = 0; kt < KDIM / 32; ++kt) {
        v16bf a = AF32 ? load_a_f32((const float*)Aptr, KDIM, rowBase, kt * 32, lane)
                       : load_a_bf16((const unsigned short*)Aptr, KDIM, rowBase, kt * 32, lane);
#pragma unroll
        for (int nt = 0; nt < 8; ++nt) {
            v16bf b = load_b(Bp, kt, ntBase + nt, lane);
            acc[nt] = __builtin_amdgcn_wmma_f32_16x16x32_bf16(
                false, a, false, b, (short)0, acc[nt], false, false);
        }
    }
    int half = lane >> 4, n = lane & 15;
#pragma unroll
    for (int nt = 0; nt < 8; ++nt) {
        int col = (ntBase + nt) * 16 + n;
        float bv = 0.f;
        if (BIAS) bv = bias[col];
#pragma unroll
        for (int r = 0; r < 8; ++r) {
            int row = rowBase + r + half * 8;       // C/D layout: VGPR r -> M = r + 8*half
            C[(size_t)row * 256 + col] = acc[nt][r] + bv;
        }
    }
}

// ---- scatter-mean pooling ---------------------------------------------------
__global__ void pool_kernel(const float* __restrict__ feat, const float* __restrict__ coords,
                            float* __restrict__ gridsum, float* __restrict__ cnt,
                            int* __restrict__ lin) {
    int t = blockIdx.x * blockDim.x + threadIdx.x;   // ROWS * 64 threads
    int p = t >> 6, sub = t & 63;
    if (p >= ROWS) return;
    int b = p >> 14;
    float cx = coords[2 * p], cy = coords[2 * p + 1];
    int ix = (int)(cx * 0.125f);  ix = ix < 0 ? 0 : (ix > 31 ? 31 : ix);   // CELL_W = 8
    int iy = (int)(cy * 0.125f);  iy = iy < 0 ? 0 : (iy > 31 ? 31 : iy);
    int l = ix * 32 + iy;
    if (sub == 0) {
        lin[p] = l;
        unsafeAtomicAdd(cnt + b * Mc + l, 1.0f);
    }
    float* dst = gridsum + ((size_t)(b * Mc + l)) * 256 + sub * 4;
    const float* src = feat + (size_t)p * 256 + sub * 4;
#pragma unroll
    for (int i = 0; i < 4; ++i) unsafeAtomicAdd(dst + i, src[i]);
}

__global__ void grid_fin(const float* __restrict__ gridsum, const float* __restrict__ cnt,
                         unsigned short* __restrict__ gbf) {
    int t = blockIdx.x * blockDim.x + threadIdx.x;   // Bn*Mc*256
    float c = cnt[t >> 8];
    c = c < 1.0f ? 1.0f : c;
    gbf[t] = f2bf(gridsum[t] / c);
}

// ---- PK/PV = pos_emb @ Wk/Wv + bias (tiny, exact f32) -----------------------
__global__ void pkv_kernel(const float* __restrict__ pos,
                           const float* __restrict__ Wk, const float* __restrict__ bk,
                           const float* __restrict__ Wv, const float* __restrict__ bv,
                           float* __restrict__ PK, float* __restrict__ PV) {
    int t = blockIdx.x * blockDim.x + threadIdx.x;   // 9*256
    if (t >= 9 * 256) return;
    int k = t >> 8, d = t & 255;
    float sk = bk[d], sv = bv[d];
    for (int c = 0; c < 256; ++c) {
        float pe = pos[k * 256 + c];
        sk = fmaf(pe, Wk[(size_t)c * 256 + d], sk);
        sv = fmaf(pe, Wv[(size_t)c * 256 + d], sv);
    }
    PK[t] = sk; PV[t] = sv;
}

// ---- tiny-window attention: one wave per point ------------------------------
__global__ void __launch_bounds__(256)
attn_kernel(const float* __restrict__ Q, const float* __restrict__ GK,
            const float* __restrict__ GV, const float* __restrict__ PK,
            const float* __restrict__ PV, const int* __restrict__ lin,
            unsigned short* __restrict__ ctx) {
    int p = blockIdx.x * (blockDim.x >> 5) + (threadIdx.x >> 5);
    if (p >= ROWS) return;
    int lane = threadIdx.x & 31;
    int b = p >> 14;
    int l = lin[p];
    int ix = l >> 5, iy = l & 31;
    const float* qp = Q + (size_t)p * 256 + lane * 8;
    float q[8];
    { v4f q0 = *(const v4f*)qp, q1 = *(const v4f*)(qp + 4);
#pragma unroll
      for (int i = 0; i < 4; ++i) { q[i] = q0[i]; q[4 + i] = q1[i]; } }

    float sc[9]; int cell[9]; bool val[9];
#pragma unroll
    for (int k = 0; k < 9; ++k) {
        int dx = k % 3 - 1, dy = k / 3 - 1;          // OFFSETS row-major over (dy,dx)
        int nx = ix + dx, ny = iy + dy;
        bool v = (nx >= 0) & (nx < 32) & (ny >= 0) & (ny < 32);
        val[k] = v; cell[k] = v ? (nx * 32 + ny) : 0;
        float partial = 0.f;
        if (v) {                                     // wave-uniform branch
            const float* gk = GK + ((size_t)(b * Mc + cell[k])) * 256 + lane * 8;
            const float* pk = PK + k * 256 + lane * 8;
#pragma unroll
            for (int j = 0; j < 8; ++j) partial = fmaf(q[j], gk[j] + pk[j], partial);
        }
#pragma unroll
        for (int m = 16; m >= 1; m >>= 1) partial += __shfl_xor(partial, m, 32);
        sc[k] = partial * 0.0625f;                   // 1/sqrt(256)
    }
    float mx = -3.4e38f;
#pragma unroll
    for (int k = 0; k < 9; ++k) if (val[k]) mx = fmaxf(mx, sc[k]);
    float w[9], se = 0.f;
#pragma unroll
    for (int k = 0; k < 9; ++k) { w[k] = val[k] ? __expf(sc[k] - mx) : 0.f; se += w[k]; }
    float inv = 1.f / se;                            // center neighbor always valid
    float a[8] = {0.f, 0.f, 0.f, 0.f, 0.f, 0.f, 0.f, 0.f};
#pragma unroll
    for (int k = 0; k < 9; ++k) {
        if (!val[k]) continue;
        float wk = w[k] * inv;
        const float* gv = GV + ((size_t)(b * Mc + cell[k])) * 256 + lane * 8;
        const float* pv = PV + k * 256 + lane * 8;
#pragma unroll
        for (int j = 0; j < 8; ++j) a[j] = fmaf(wk, gv[j] + pv[j], a[j]);
    }
    u16x8 o;
#pragma unroll
    for (int j = 0; j < 8; ++j) o[j] = f2bf(a[j]);
    *(u16x8*)(ctx + (size_t)p * 256 + lane * 8) = o;
}

// ---- LayerNorm + residual + mask select: one wave per row -------------------
// out[row] = mask ? LN(feat[row] + h[row]) * gamma + beta : feat[row]
__global__ void __launch_bounds__(256)
ln_kernel(const float* __restrict__ h, const float* __restrict__ feat,
          const unsigned char* __restrict__ vmask, const float* __restrict__ gamma,
          const float* __restrict__ beta, float* __restrict__ out) {
    int row = blockIdx.x * (blockDim.x >> 5) + (threadIdx.x >> 5);
    if (row >= ROWS) return;
    int lane = threadIdx.x & 31;
    const float* hp = h    + (size_t)row * 256 + lane * 8;
    const float* fp = feat + (size_t)row * 256 + lane * 8;
    v4f h0 = *(const v4f*)hp, h1 = *(const v4f*)(hp + 4);
    v4f f0 = *(const v4f*)fp, f1 = *(const v4f*)(fp + 4);
    float hv[8], fv[8];
    float s = 0.f, sq = 0.f;
#pragma unroll
    for (int i = 0; i < 8; ++i) {
        fv[i] = (i < 4) ? f0[i] : f1[i - 4];
        hv[i] = fv[i] + ((i < 4) ? h0[i] : h1[i - 4]);
        s += hv[i];
        sq = fmaf(hv[i], hv[i], sq);
    }
#pragma unroll
    for (int m = 1; m < 32; m <<= 1) {               // full 32-lane reduction
        s  += __shfl_xor(s,  m, 32);
        sq += __shfl_xor(sq, m, 32);
    }
    float mu = s * (1.f / 256.f);
    float var = sq * (1.f / 256.f) - mu * mu;
    float rs = rsqrtf(var + 1e-5f);
    bool mv = vmask[row] != 0;
    const float* gp = gamma + lane * 8;
    const float* bp = beta  + lane * 8;
    v4f g0 = *(const v4f*)gp, g1 = *(const v4f*)(gp + 4);
    v4f b0 = *(const v4f*)bp, b1 = *(const v4f*)(bp + 4);
    v4f o0, o1;
#pragma unroll
    for (int i = 0; i < 4; ++i) {
        o0[i] = mv ? (hv[i]     - mu) * rs * g0[i] + b0[i] : fv[i];
        o1[i] = mv ? (hv[4 + i] - mu) * rs * g1[i] + b1[i] : fv[4 + i];
    }
    float* op = out + (size_t)row * 256 + lane * 8;
    *(v4f*)op       = o0;
    *(v4f*)(op + 4) = o1;
}

// ---- launcher ---------------------------------------------------------------
extern "C" void kernel_launch(void* const* d_in, const int* in_sizes, int n_in,
                              void* d_out, int out_size, void* d_ws, size_t ws_size,
                              hipStream_t stream) {
    const float* features = (const float*)d_in[0];
    const float* coords   = (const float*)d_in[1];
    const unsigned char* vmask = (const unsigned char*)d_in[2];
    const float* Wf = (const float*)d_in[3];  const float* bf = (const float*)d_in[4];
    const float* Wq = (const float*)d_in[5];  const float* bq = (const float*)d_in[6];
    const float* Wk = (const float*)d_in[7];  const float* bk = (const float*)d_in[8];
    const float* Wv = (const float*)d_in[9];  const float* bv = (const float*)d_in[10];
    const float* Wo = (const float*)d_in[11]; const float* bo = (const float*)d_in[12];
    const float* pos   = (const float*)d_in[13];
    const float* gamma = (const float*)d_in[14];
    const float* beta  = (const float*)d_in[15];

    char* ws = (char*)d_ws;
    size_t cur = 0;
    auto carve = [&](size_t bytes) -> char* {
        char* p = ws + cur;
        cur = (cur + bytes + 255) & ~(size_t)255;
        return p;
    };
    float*          feat    = (float*)carve((size_t)ROWS * 256 * 4);      // 64 MiB
    float*          Qbuf    = (float*)carve((size_t)ROWS * 256 * 4);      // 64 MiB (Q, then h)
    unsigned short* ctx     = (unsigned short*)carve((size_t)ROWS * 256 * 2);
    float*          gridsum = (float*)carve((size_t)Bn * Mc * 256 * 4);
    float*          cnt     = (float*)carve((size_t)Bn * Mc * 4);
    unsigned short* gridbf  = (unsigned short*)carve((size_t)Bn * Mc * 256 * 2);
    float*          GK      = (float*)carve((size_t)Bn * Mc * 256 * 4);
    float*          GV      = (float*)carve((size_t)Bn * Mc * 256 * 4);
    float*          PK      = (float*)carve(9 * 256 * 4);
    float*          PV      = (float*)carve(9 * 256 * 4);
    int*            lin     = (int*)carve((size_t)ROWS * 4);
    unsigned short* Wfp     = (unsigned short*)carve((size_t)FINc * 256 * 2);
    unsigned short* Wqp     = (unsigned short*)carve((size_t)256 * 256 * 2);
    unsigned short* Wkp     = (unsigned short*)carve((size_t)256 * 256 * 2);
    unsigned short* Wvp     = (unsigned short*)carve((size_t)256 * 256 * 2);
    unsigned short* Wop     = (unsigned short*)carve((size_t)256 * 256 * 2);

    hipMemsetAsync(gridsum, 0, (size_t)Bn * Mc * 256 * 4, stream);
    hipMemsetAsync(cnt,     0, (size_t)Bn * Mc * 4, stream);

    pack_w<<<(FINc * 256) / 256, 256, 0, stream>>>(Wf, Wfp, FINc, 256);
    pack_w<<<(256 * 256) / 256, 256, 0, stream>>>(Wq, Wqp, 256, 256);
    pack_w<<<(256 * 256) / 256, 256, 0, stream>>>(Wk, Wkp, 256, 256);
    pack_w<<<(256 * 256) / 256, 256, 0, stream>>>(Wv, Wvp, 256, 256);
    pack_w<<<(256 * 256) / 256, 256, 0, stream>>>(Wo, Wop, 256, 256);
    pkv_kernel<<<9, 256, 0, stream>>>(pos, Wk, bk, Wv, bv, PK, PV);

    dim3 gBig(ROWS / 16 / 8, 2);          // 16-row waves x 2 column halves
    dim3 gGrid(Bn * Mc / 16 / 8, 2);

    // feat = features @ Wf + bf           (65536 x 128 x 256, WMMA bf16)
    gemm256<FINc, true, true><<<gBig, 256, 0, stream>>>(features, Wfp, bf, feat, ROWS);
    // scatter-mean pooling into the 32x32 grid
    pool_kernel<<<(ROWS * 64) / 256, 256, 0, stream>>>(feat, coords, gridsum, cnt, lin);
    grid_fin<<<(Bn * Mc * 256) / 256, 256, 0, stream>>>(gridsum, cnt, gridbf);
    // GK/GV = grid @ Wk / Wv              (4096 x 256 x 256 each, WMMA bf16)
    gemm256<256, false, false><<<gGrid, 256, 0, stream>>>(gridbf, Wkp, nullptr, GK, Bn * Mc);
    gemm256<256, false, false><<<gGrid, 256, 0, stream>>>(gridbf, Wvp, nullptr, GV, Bn * Mc);
    // Q = feat @ Wq + bq                  (65536 x 256 x 256, WMMA bf16)
    gemm256<256, true, true><<<gBig, 256, 0, stream>>>(feat, Wqp, bq, Qbuf, ROWS);
    // per-point 9-neighbor attention (wave32 per point); consumes Q
    attn_kernel<<<ROWS / 8, 256, 0, stream>>>(Qbuf, GK, GV, PK, PV, lin, ctx);
    // h = ctx @ Wo + bo  (reuse Qbuf slot), then fused residual+LN+mask select
    gemm256<256, false, true><<<gBig, 256, 0, stream>>>(ctx, Wop, bo, Qbuf, ROWS);
    ln_kernel<<<ROWS / 8, 256, 0, stream>>>(Qbuf, feat, vmask, gamma, beta, (float*)d_out);
}
